// WindowAttention_26190710571074
// MI455X (gfx1250) — matrix operand order repaired and checked by reading, not compiled, not run on hardware
//
#include <hip/hip_runtime.h>
#include <hip/hip_bf16.h>

#define NTOK  49
#define NP    64
#define DIM   128
#define NHEAD 4
#define HD    32
#define NWIN  256
#define NWGBLK 4096

typedef __attribute__((ext_vector_type(16))) _Float16 v16h;
typedef __attribute__((ext_vector_type(8)))  _Float16 v8h;
typedef __attribute__((ext_vector_type(8)))  float    v8f;
typedef __attribute__((ext_vector_type(4)))  unsigned int u32x4;

// ---------------------------------------------------------------------------
// Fragment loaders matching CDNA5 WMMA VGPR layouts (CDNA5 ISA):
// A 16x32 f16: lane(0-15)=row M, halves K = {hi*8..hi*8+7} U {16+hi*8..+7}
// B 32x16 f16: lane&15 = col N, halves K = {hi*8..hi*8+7} U {16+hi*8..+7}
//   -> B-side matrices stored transposed [N][K] so this is two 16B loads.
// C/D f32 16x16: element (M = 8*hi + r, N = lane&15) in VGPR slot r.
// ---------------------------------------------------------------------------
__device__ __forceinline__ v16h load_a16(const _Float16* row, int k0, int hi) {
    union { v16h v; u32x4 q[2]; } u;
    u.q[0] = *(const u32x4*)(row + k0 + hi * 8);
    u.q[1] = *(const u32x4*)(row + k0 + 16 + hi * 8);
    return u.v;
}
__device__ __forceinline__ v16h load_b16(const _Float16* row, int k0, int hi) {
    union { v16h v; u32x4 q[2]; } u;
    u.q[0] = *(const u32x4*)(row + k0 + hi * 8);
    u.q[1] = *(const u32x4*)(row + k0 + 16 + hi * 8);
    return u.v;
}
__device__ __forceinline__ v8f wmma16(v16h a, v16h b, v8f c) {
    return __builtin_amdgcn_wmma_f32_16x16x32_f16(false, a, false, b,
                                                  (short)0, c, false, false);
}

// ---------------------------------------------------------------------------
// Prep kernel: f32->f16 weight transpose-convert + padded rel-pos bias table
// ---------------------------------------------------------------------------
__global__ void winattn_prep(const float* __restrict__ qkv_w,
                             const float* __restrict__ proj_w,
                             const float* __restrict__ bias_table,
                             _Float16* __restrict__ wqkvt,   // [384][128]
                             _Float16* __restrict__ wprojt,  // [128][128]
                             float* __restrict__ biasp)      // [4][64][64]
{
    int idx = blockIdx.x * 256 + threadIdx.x;
    const int A = 384 * 128, B = 128 * 128, C = NHEAD * NP * NP;
    if (idx < A) {
        int o = idx / 128, i = idx - o * 128;
        wqkvt[o * 128 + i] = (_Float16)qkv_w[i * 384 + o];
    } else if (idx < A + B) {
        int t = idx - A;
        int o = t / 128, i = t - o * 128;
        wprojt[o * 128 + i] = (_Float16)proj_w[i * 128 + o];
    } else if (idx < A + B + C) {
        int t = idx - A - B;
        int h = t >> 12, r = t & 4095;
        int i = r >> 6, j = r & 63;
        float v = 0.f;
        if (i < NTOK && j < NTOK) {
            int yi = i / 7, xi = i - yi * 7, yj = j / 7, xj = j - yj * 7;
            int rel = (yi - yj + 6) * 13 + (xi - xj + 6);
            v = bias_table[rel * NHEAD + h];
        }
        biasp[t] = v;
    }
}

// ---------------------------------------------------------------------------
// Main kernel: one workgroup (8 wave32s) per window
// ---------------------------------------------------------------------------
__global__ __launch_bounds__(256, 1) void winattn_main(
    const float* __restrict__ x,        // [4096][49][128]
    const float* __restrict__ mask,     // [256][49][49]
    const float* __restrict__ qkv_b,    // [384]
    const float* __restrict__ proj_b,   // [128]
    const _Float16* __restrict__ wqkvt, // [384][128] (f16, transposed)
    const _Float16* __restrict__ wprojt,// [128][128] (f16, transposed)
    const float* __restrict__ biasp,    // [4][64][64]
    float* __restrict__ out)            // [4096][49][128]
{
    // LDS (144 KB total; ISA allows up to 320 KB/workgroup)
    __shared__ float    SX[NHEAD * NP * NP];   // 64KB: x-tile (f16) then scores (f32)
    __shared__ _Float16 QKO[NP * 256];         // 32KB: Q|K f16, later attn-out tile
    __shared__ _Float16 Vt[NHEAD * HD * NP];   // 16KB: V transposed [h*32+d][tok]
    __shared__ _Float16 Pb[NHEAD * NP * NP];   // 32KB: softmax probs f16

    const int tid  = threadIdx.x;
    const int lane = tid & 31;
    const int wv   = tid >> 5;      // wave id 0..7
    const int ln   = lane & 15;
    const int hi   = lane >> 4;
    const int b    = blockIdx.x;

    const v8f vzero = {0.f, 0.f, 0.f, 0.f, 0.f, 0.f, 0.f, 0.f};
    _Float16* Xs = (_Float16*)SX;   // [NP][DIM], dead after phase 2

    // ---- phase 1: x -> LDS f16 (zero-pad rows 49..63); prefetch weights --
    {
        // Warm L2/WGP$ with the f16 weights while x streams in.
        if (tid < 96) __builtin_prefetch(wqkvt + tid * 512, 0, 1);
        else if (tid < 128) __builtin_prefetch(wprojt + (tid - 96) * 512, 0, 1);

        const float* xw = x + (size_t)b * NTOK * DIM;
        for (int idx = tid; idx < NP * DIM; idx += 256) {
            int m = idx >> 7;
            float v = (m < NTOK) ? xw[idx] : 0.f;
            Xs[idx] = (_Float16)v;
        }
    }
    __syncthreads();

    // ---- phase 2: QKV GEMM  [64x128] @ [128x384], 3 N-tiles per wave -----
    for (int t = 0; t < 3; ++t) {
        int nt = wv * 3 + t;
        int nb = nt * 16;
        const _Float16* brow = wqkvt + (nb + ln) * DIM;
        v8f acc[4] = {vzero, vzero, vzero, vzero};
        for (int kt = 0; kt < 4; ++kt) {
            int k0 = kt * 32;
            v16h bf = load_b16(brow, k0, hi);
            #pragma unroll
            for (int mt = 0; mt < 4; ++mt) {
                v16h af = load_a16(Xs + (mt * 16 + ln) * DIM, k0, hi);
                acc[mt] = wmma16(af, bf, acc[mt]);
            }
        }
        int ncol = nb + ln;
        float bn = qkv_b[ncol];
        if (nt < 16) {
            // Q|K tile: scattered rows, per-element f16 stores
            #pragma unroll
            for (int mt = 0; mt < 4; ++mt)
                #pragma unroll
                for (int r = 0; r < 8; ++r)
                    QKO[(mt * 16 + hi * 8 + r) * 256 + ncol] =
                        (_Float16)(acc[mt][r] + bn);
        } else {
            // V tile: lane's 8 elements are contiguous tokens in V^T row ->
            // pack to f16 and store 16B at once (ds_store_b128)
            int vc = ncol - 256;            // = h*32 + d
            #pragma unroll
            for (int mt = 0; mt < 4; ++mt) {
                union { v8h h; u32x4 q; } pk;
                #pragma unroll
                for (int r = 0; r < 8; ++r)
                    pk.h[r] = (_Float16)(acc[mt][r] + bn);
                *(u32x4*)(Vt + vc * NP + mt * 16 + hi * 8) = pk.q;
            }
        }
    }
    __syncthreads();

    // ---- phase 3: S = Q K^T (2 waves per head, 2 N-tiles each) -----------
    {
        const int h = wv >> 1;
        const int half = wv & 1;
        float* Sh = SX + h * NP * NP;
        for (int u = 0; u < 2; ++u) {
            int nt = half * 2 + u;
            v16h bf = load_b16(QKO + (nt * 16 + ln) * 256 + DIM + h * HD, 0, hi);
            #pragma unroll
            for (int mt = 0; mt < 4; ++mt) {
                v16h af = load_a16(QKO + (mt * 16 + ln) * 256 + h * HD, 0, hi);
                v8f acc = wmma16(af, bf, vzero);
                #pragma unroll
                for (int r = 0; r < 8; ++r)
                    Sh[(mt * 16 + hi * 8 + r) * NP + nt * 16 + ln] = acc[r];
            }
        }
    }
    __syncthreads();

    // ---- phase 4: softmax rows (+scale +bias +mask), 1 row per thread ----
    {
        const float scale = 0.17677669529663689f;   // 1/sqrt(32)
        const int widx = b & (NWIN - 1);
        const int h = tid >> 6, i = tid & 63;
        float*    Sr = SX + (h * NP + i) * NP;
        _Float16* Pr = Pb + (h * NP + i) * NP;
        if (i < NTOK) {
            const float* br = biasp + (h * NP + i) * NP;
            const float* mr = mask + ((size_t)widx * NTOK + i) * NTOK;
            float mx = -3.0e38f;
            for (int j = 0; j < NTOK; ++j) {
                float v = Sr[j] * scale + br[j] + mr[j];
                Sr[j] = v;
                mx = fmaxf(mx, v);
            }
            float sum = 0.f;
            for (int j = 0; j < NTOK; ++j) {
                float e = __expf(Sr[j] - mx);
                Sr[j] = e;
                sum += e;
            }
            float inv = 1.f / sum;
            for (int j = 0; j < NTOK; ++j) Pr[j] = (_Float16)(Sr[j] * inv);
            for (int j = NTOK; j < NP; ++j) Pr[j] = (_Float16)0.f;
        } else {
            for (int j = 0; j < NP; ++j) Pr[j] = (_Float16)0.f;
        }
    }
    __syncthreads();

    // ---- phase 5: O = P V (2 waves per head, 1 d-tile each) --------------
    {
        const int h = wv >> 1;
        const int nt = wv & 1;
        _Float16* Outs = QKO;                     // [NP][DIM], aliases dead Q|K
        const _Float16* Ph = Pb + h * NP * NP;
        v8f acc[4] = {vzero, vzero, vzero, vzero};
        for (int kt = 0; kt < 2; ++kt) {
            int k0 = kt * 32;
            v16h bf = load_b16(Vt + (h * HD + nt * 16 + ln) * NP, k0, hi);
            #pragma unroll
            for (int mt = 0; mt < 4; ++mt) {
                v16h af = load_a16(Ph + (mt * 16 + ln) * NP, k0, hi);
                acc[mt] = wmma16(af, bf, acc[mt]);
            }
        }
        #pragma unroll
        for (int mt = 0; mt < 4; ++mt)
            #pragma unroll
            for (int r = 0; r < 8; ++r)
                Outs[(mt * 16 + hi * 8 + r) * DIM + h * HD + nt * 16 + ln] =
                    (_Float16)acc[mt][r];
    }
    __syncthreads();

    // ---- phase 6: projection [64x128] @ [128x128], 1 N-tile per wave -----
    {
        const _Float16* Outs = QKO;
        float* orow = out + (size_t)b * NTOK * DIM;
        const int nb = wv * 16;
        const _Float16* brow = wprojt + (nb + ln) * DIM;
        v8f acc[4] = {vzero, vzero, vzero, vzero};
        for (int kt = 0; kt < 4; ++kt) {
            int k0 = kt * 32;
            v16h bf = load_b16(brow, k0, hi);
            #pragma unroll
            for (int mt = 0; mt < 4; ++mt) {
                v16h af = load_a16(Outs + (mt * 16 + ln) * DIM, k0, hi);
                acc[mt] = wmma16(af, bf, acc[mt]);
            }
        }
        float bn = proj_b[nb + ln];
        #pragma unroll
        for (int mt = 0; mt < 4; ++mt)
            #pragma unroll
            for (int r = 0; r < 8; ++r) {
                int m = mt * 16 + hi * 8 + r;
                if (m < NTOK) orow[m * DIM + nb + ln] = acc[mt][r] + bn;
            }
    }
}

// ---------------------------------------------------------------------------
extern "C" void kernel_launch(void* const* d_in, const int* in_sizes, int n_in,
                              void* d_out, int out_size, void* d_ws, size_t ws_size,
                              hipStream_t stream) {
    const float* x      = (const float*)d_in[0];
    const float* mask   = (const float*)d_in[1];
    const float* qkv_w  = (const float*)d_in[2];
    const float* qkv_b  = (const float*)d_in[3];
    const float* proj_w = (const float*)d_in[4];
    const float* proj_b = (const float*)d_in[5];
    const float* btab   = (const float*)d_in[6];

    char* ws = (char*)d_ws;
    _Float16* wqkvt  = (_Float16*)ws;                      // 384*128*2 = 98304 B
    _Float16* wprojt = (_Float16*)(ws + 98304);            // 128*128*2 = 32768 B
    float*    biasp  = (float*)(ws + 98304 + 32768);       // 4*64*64*4 = 65536 B

    const int prep_items = 384 * 128 + 128 * 128 + NHEAD * NP * NP;
    winattn_prep<<<(prep_items + 255) / 256, 256, 0, stream>>>(
        qkv_w, proj_w, btab, wqkvt, wprojt, biasp);

    winattn_main<<<NWGBLK, 256, 0, stream>>>(
        x, mask, qkv_b, proj_b, wqkvt, wprojt, biasp, (float*)d_out);
}
